// Conv_4432406250054
// MI455X (gfx1250) — matrix operand order repaired and checked by reading, not compiled, not run on hardware
//
#include <hip/hip_runtime.h>
#include <hip/hip_bf16.h>
#include <math.h>

// ---------------------------------------------------------------------------
// CDNA5 / gfx1250 implementation.  wave32, WMMA 16x16x32 f16 for the MLP GEMMs,
// GLOBAL_LOAD_ASYNC_TO_LDS for conv-weight staging (ASYNCcnt path).
// ---------------------------------------------------------------------------

typedef __attribute__((ext_vector_type(16))) _Float16 v16h;
typedef __attribute__((ext_vector_type(8)))  float    v8f;
typedef __attribute__((ext_vector_type(4)))  int      v4i;

#define NPIX   65536      // 4*128*128
#define HW     16384      // 128*128
#define CDIM   128
#define HID    512
#define GELU_K 0.70710678118654752f

#ifndef __has_builtin
#define __has_builtin(x) 0
#endif
#if defined(__HIP_DEVICE_COMPILE__) && \
    __has_builtin(__builtin_amdgcn_global_load_async_to_lds_b128) && \
    __has_builtin(__builtin_amdgcn_global_load_async_to_lds_b32)
#define HAVE_ASYNC_LDS 1
#else
#define HAVE_ASYNC_LDS 0
#endif

// ---------------------------------------------------------------------------
// Kernel 1: fused spatial convs -> shuffled y in [pix][128] layout.
// ---------------------------------------------------------------------------
__global__ __launch_bounds__(256) void spatial_kernel(
    const float* __restrict__ x,
    const float* __restrict__ offw_g, const float* __restrict__ offb_g,
    const float* __restrict__ defw_g,
    const float* __restrict__ dww_g,  const float* __restrict__ dwb_g,
    const float* __restrict__ dw2w_g, const float* __restrict__ dw2b_g,
    float* __restrict__ yraw)
{
    __shared__ __attribute__((aligned(16))) float offw[18 * 32 * 9]; // 20.25 KB
    __shared__ __attribute__((aligned(16))) float offb[20];
    __shared__ __attribute__((aligned(16))) float defw[32 * 9];
    __shared__ __attribute__((aligned(16))) float dww[64 * 49];      // 12.25 KB
    __shared__ __attribute__((aligned(16))) float dwb[64];
    __shared__ __attribute__((aligned(16))) float dw2b[32];

    const int tid = threadIdx.x;
#if HAVE_ASYNC_LDS
    // Async DMA of the weight tables into LDS (ASYNCcnt path).
    for (int i = tid; i < 1296; i += 256)   // 20736 B of offw, 16 B per op
        __builtin_amdgcn_global_load_async_to_lds_b128(
            (v4i*)(offw_g + i * 4), (v4i*)(offw + i * 4), 0, 0);
    for (int i = tid; i < 784; i += 256)    // 12544 B of dww
        __builtin_amdgcn_global_load_async_to_lds_b128(
            (v4i*)(dww_g + i * 4), (v4i*)(dww + i * 4), 0, 0);
    for (int i = tid; i < 72; i += 256)     // 1152 B of defw
        __builtin_amdgcn_global_load_async_to_lds_b128(
            (v4i*)(defw_g + i * 4), (v4i*)(defw + i * 4), 0, 0);
    if (tid < 16)                           // 256 B of dwb
        __builtin_amdgcn_global_load_async_to_lds_b128(
            (v4i*)(dwb_g + tid * 4), (v4i*)(dwb + tid * 4), 0, 0);
    if (tid < 8)                            // 128 B of dw2b
        __builtin_amdgcn_global_load_async_to_lds_b128(
            (v4i*)(dw2b_g + tid * 4), (v4i*)(dw2b + tid * 4), 0, 0);
    if (tid < 18)                           // 72 B of offb, 4 B per op
        __builtin_amdgcn_global_load_async_to_lds_b32(
            (int*)(offb_g + tid), (int*)(offb + tid), 0, 0);
    asm volatile("s_wait_asynccnt 0" ::: "memory");
#else
    for (int i = tid; i < 18 * 32 * 9; i += 256) offw[i] = offw_g[i];
    for (int i = tid; i < 18;          i += 256) offb[i] = offb_g[i];
    for (int i = tid; i < 288;         i += 256) defw[i] = defw_g[i];
    for (int i = tid; i < 64 * 49;     i += 256) dww[i]  = dww_g[i];
    for (int i = tid; i < 64;          i += 256) dwb[i]  = dwb_g[i];
    for (int i = tid; i < 32;          i += 256) dw2b[i] = dw2b_g[i];
#endif
    __syncthreads();

    const int pix = blockIdx.x * 256 + tid;
    const int b   = pix >> 14;
    const int hw  = pix & (HW - 1);
    const int h   = hw >> 7;
    const int w   = hw & 127;
    const float* xb = x + (size_t)b * CDIM * HW;
    float* yo = yraw + (size_t)pix * CDIM;

    // ---- offset conv: 18 = conv3x3(x[0:32]) ----
    float off[18];
#pragma unroll
    for (int oc = 0; oc < 18; ++oc) off[oc] = offb[oc];
    for (int ky = 0; ky < 3; ++ky) {
        const int yy = h + ky - 1;
        if ((unsigned)yy >= 128u) continue;
        for (int kx = 0; kx < 3; ++kx) {
            const int xx = w + kx - 1;
            if ((unsigned)xx >= 128u) continue;
            const int kk = ky * 3 + kx;
            for (int ic = 0; ic < 32; ++ic) {
                const float xv = xb[(size_t)ic * HW + yy * 128 + xx];
#pragma unroll
                for (int oc = 0; oc < 18; ++oc)
                    off[oc] = fmaf(xv, offw[(oc * 32 + ic) * 9 + kk], off[oc]);
            }
        }
    }

    // ---- deformable depthwise, channels 0..31 ----
    float acc0[32];
#pragma unroll
    for (int c = 0; c < 32; ++c) acc0[c] = 0.0f;
#pragma unroll
    for (int k = 0; k < 9; ++k) {
        const float ys = (float)h - 1.0f + (float)(k / 3) + off[2 * k];
        const float xs = (float)w - 1.0f + (float)(k % 3) + off[2 * k + 1];
        const float y0 = floorf(ys), x0 = floorf(xs);
#pragma unroll
        for (int dy = 0; dy < 2; ++dy) {
#pragma unroll
            for (int dx = 0; dx < 2; ++dx) {
                const float yc = y0 + (float)dy, xc = x0 + (float)dx;
                float wgt = (1.0f - fabsf(ys - yc)) * (1.0f - fabsf(xs - xc));
                const bool valid = (yc >= 0.0f) && (yc <= 127.0f) &&
                                   (xc >= 0.0f) && (xc <= 127.0f);
                if (!valid) wgt = 0.0f;
                const int yi = (int)fminf(fmaxf(yc, 0.0f), 127.0f);
                const int xi = (int)fminf(fmaxf(xc, 0.0f), 127.0f);
                const float* cp = xb + yi * 128 + xi;
                for (int c = 0; c < 32; ++c) {
                    const float g = cp[(size_t)c * HW];
                    acc0[c] = fmaf(g * wgt, defw[c * 9 + k], acc0[c]);
                }
            }
        }
    }
#pragma unroll
    for (int c = 0; c < 32; ++c) yo[2 * c] = acc0[c];   // shuffle: orig o<64 -> 2o

    // ---- depthwise 7x7, channels 32..95 ----
    for (int c = 0; c < 64; ++c) {
        float acc = dwb[c];
        for (int ky = 0; ky < 7; ++ky) {
            const int yy = h + ky - 3;
            if ((unsigned)yy >= 128u) continue;
            for (int kx = 0; kx < 7; ++kx) {
                const int xx = w + kx - 3;
                if ((unsigned)xx >= 128u) continue;
                acc = fmaf(xb[(size_t)(32 + c) * HW + yy * 128 + xx],
                           dww[c * 49 + ky * 7 + kx], acc);
            }
        }
        const int o  = 32 + c;
        const int co = (o < 64) ? (2 * o) : (2 * (o - 64) + 1);
        yo[co] = acc;
    }

    // ---- dense 3x3 dilation-2 conv, channels 96..127 ----
    float acc2[32];
#pragma unroll
    for (int oc = 0; oc < 32; ++oc) acc2[oc] = dw2b[oc];
    for (int ky = 0; ky < 3; ++ky) {
        const int yy = h + 2 * ky - 2;
        if ((unsigned)yy >= 128u) continue;
        for (int kx = 0; kx < 3; ++kx) {
            const int xx = w + 2 * kx - 2;
            if ((unsigned)xx >= 128u) continue;
            const int kk = ky * 3 + kx;
            for (int ic = 0; ic < 32; ++ic) {
                const float xv = xb[(size_t)(96 + ic) * HW + yy * 128 + xx];
#pragma unroll
                for (int oc = 0; oc < 32; ++oc)
                    acc2[oc] = fmaf(xv, dw2w_g[(oc * 32 + ic) * 9 + kk], acc2[oc]);
            }
        }
    }
#pragma unroll
    for (int oc = 0; oc < 32; ++oc)
        yo[2 * (96 + oc - 64) + 1] = acc2[oc];          // orig >=64 -> 2(o-64)+1
}

// ---------------------------------------------------------------------------
// BatchNorm statistics -> folded per-channel scale/shift. One block per channel.
// ---------------------------------------------------------------------------
__global__ __launch_bounds__(256) void bnstats_kernel(
    const float* __restrict__ data, int C, int npix,
    const float* __restrict__ g, const float* __restrict__ bta,
    float* __restrict__ scale, float* __restrict__ shift)
{
    __shared__ float ssum[256];
    __shared__ float ssq[256];
    const int c = blockIdx.x;
    const int tid = threadIdx.x;
    float s = 0.0f, q = 0.0f;
    for (int p = tid; p < npix; p += 256) {
        const float v = data[(size_t)p * C + c];
        s += v;
        q = fmaf(v, v, q);
    }
    ssum[tid] = s; ssq[tid] = q;
    __syncthreads();
    for (int o = 128; o > 0; o >>= 1) {
        if (tid < o) { ssum[tid] += ssum[tid + o]; ssq[tid] += ssq[tid + o]; }
        __syncthreads();
    }
    if (tid == 0) {
        const float inv_n = 1.0f / (float)npix;
        const float mu  = ssum[0] * inv_n;
        const float var = ssq[0] * inv_n - mu * mu;
        const float is  = rsqrtf(var + 1e-5f);
        const float sc  = g[c] * is;
        scale[c] = sc;
        shift[c] = bta[c] - mu * sc;
    }
}

// ---------------------------------------------------------------------------
// GEMM1: t = gelu( bn1(y)[65536,128] @ w1[128,512] + b1 ), f16 output.
// Block tile 32(M) x 256(N): waves 2(M) x 4(N); K = 128 staged in 2 halves.
// Grid (2048, 2): activation tensor read only 2x.
// ---------------------------------------------------------------------------
__global__ __launch_bounds__(256) void gemm1_kernel(
    const float* __restrict__ yraw,
    const float* __restrict__ sc1, const float* __restrict__ sh1,
    const float* __restrict__ w1,  const float* __restrict__ b1,
    _Float16* __restrict__ t)
{
    __shared__ __attribute__((aligned(32))) _Float16 lB[256 * 64];  // [col][k%64] 32KB
    __shared__ float s_sc[128];
    __shared__ float s_sh[128];

    const int tid    = threadIdx.x;
    const int rowblk = blockIdx.x * 32;
    const int colblk = blockIdx.y * 256;
    for (int i = tid; i < 128; i += 256) { s_sc[i] = sc1[i]; s_sh[i] = sh1[i]; }

    const int wave  = tid >> 5, lane = tid & 31;
    const int mwave = wave >> 2, nwave = wave & 3;
    const int rowbase = rowblk + mwave * 16;
    const int arow    = rowbase + (lane & 15);
    const int khalf   = (lane < 16) ? 0 : 16;   // A fragment K base per half-wave
    const int kbh     = (lane < 16) ? 0 : 16;   // B fragment K base per half-wave

    const float* ap = yraw + (size_t)arow * CDIM;
    __builtin_prefetch(ap, 0, 3);               // global_prefetch_b8

    v8f acc[4];
#pragma unroll
    for (int n = 0; n < 4; ++n)
#pragma unroll
        for (int r = 0; r < 8; ++r) acc[n][r] = 0.0f;

    for (int kh = 0; kh < 2; ++kh) {
        __syncthreads();
        for (int i = tid; i < 256 * 64; i += 256) {
            const int col = i >> 6, k = i & 63;
            lB[i] = (_Float16)w1[(size_t)(kh * 64 + k) * HID + colblk + col];
        }
        __syncthreads();
#pragma unroll
        for (int ks = 0; ks < 2; ++ks) {
            const int kc = kh * 64 + ks * 32 + khalf;
            v16h a;
#pragma unroll
            for (int i = 0; i < 8; ++i) {
                const int k = kc + i;
                a[i] = (_Float16)fmaf(ap[k], s_sc[k], s_sh[k]);
            }
#pragma unroll
            for (int i = 0; i < 8; ++i) {
                const int k = kc + 8 + i;
                a[8 + i] = (_Float16)fmaf(ap[k], s_sc[k], s_sh[k]);
            }
            const int kb = ks * 32 + kbh;
            v16h bf[4];
#pragma unroll
            for (int n = 0; n < 4; ++n) {
                const int col = nwave * 64 + n * 16 + (lane & 15);
                bf[n] = *(const v16h*)&lB[col * 64 + kb];
            }
#pragma unroll
            for (int n = 0; n < 4; ++n)
                acc[n] = __builtin_amdgcn_wmma_f32_16x16x32_f16(
                    false, a, false, bf[n], (short)0, acc[n], false, false);
        }
    }

    const int crow0 = rowbase + ((lane >> 4) << 3);
#pragma unroll
    for (int n = 0; n < 4; ++n) {
        const int col  = colblk + nwave * 64 + n * 16 + (lane & 15);
        const float bs = b1[col];
#pragma unroll
        for (int r = 0; r < 8; ++r) {
            float v = acc[n][r] + bs;
            v = 0.5f * v * (1.0f + erff(v * GELU_K));
            t[(size_t)(crow0 + r) * HID + col] = (_Float16)v;
        }
    }
}

// ---------------------------------------------------------------------------
// GEMM2: z = t[65536,512](f16) @ w2[512,128] + b2, f32 output.
// Block tile 64(M) x 128(N = full): waves 4(M) x 2(N); K = 512 in 4 quarters.
// Grid (1024, 1): hidden tensor read exactly once.
// ---------------------------------------------------------------------------
__global__ __launch_bounds__(256) void gemm2_kernel(
    const _Float16* __restrict__ t,
    const float* __restrict__ w2, const float* __restrict__ b2,
    float* __restrict__ z)
{
    __shared__ __attribute__((aligned(32))) _Float16 lB[128 * 128]; // [col][k%128] 32KB

    const int tid    = threadIdx.x;
    const int rowblk = blockIdx.x * 64;
    const int wave   = tid >> 5, lane = tid & 31;
    const int mwave  = wave >> 1, nwave = wave & 1;
    const int rowbase = rowblk + mwave * 16;
    const int arow    = rowbase + (lane & 15);
    const int khalf   = (lane < 16) ? 0 : 16;
    const int kbh     = (lane < 16) ? 0 : 16;

    const _Float16* ap = t + (size_t)arow * HID;
    __builtin_prefetch(ap, 0, 3);

    v8f acc[4];
#pragma unroll
    for (int n = 0; n < 4; ++n)
#pragma unroll
        for (int r = 0; r < 8; ++r) acc[n][r] = 0.0f;

    for (int kq = 0; kq < 4; ++kq) {
        __syncthreads();
        for (int i = tid; i < 128 * 128; i += 256) {
            const int col = i >> 7, k = i & 127;
            lB[i] = (_Float16)w2[(size_t)(kq * 128 + k) * CDIM + col];
        }
        __syncthreads();
#pragma unroll
        for (int ks = 0; ks < 4; ++ks) {
            const int kc = kq * 128 + ks * 32 + khalf;
            v16h a;
#pragma unroll
            for (int i = 0; i < 8; ++i) a[i]     = ap[kc + i];
#pragma unroll
            for (int i = 0; i < 8; ++i) a[8 + i] = ap[kc + 8 + i];
            const int kb = ks * 32 + kbh;
            v16h bf[4];
#pragma unroll
            for (int n = 0; n < 4; ++n) {
                const int col = nwave * 64 + n * 16 + (lane & 15);
                bf[n] = *(const v16h*)&lB[col * 128 + kb];
            }
#pragma unroll
            for (int n = 0; n < 4; ++n)
                acc[n] = __builtin_amdgcn_wmma_f32_16x16x32_f16(
                    false, a, false, bf[n], (short)0, acc[n], false, false);
        }
    }

    const int crow0 = rowbase + ((lane >> 4) << 3);
#pragma unroll
    for (int n = 0; n < 4; ++n) {
        const int col  = nwave * 64 + n * 16 + (lane & 15);
        const float bs = b2[col];
#pragma unroll
        for (int r = 0; r < 8; ++r)
            z[(size_t)(crow0 + r) * CDIM + col] = acc[n][r] + bs;
    }
}

// ---------------------------------------------------------------------------
// Final: out = gelu(x + bn2(z)), NCHW output.
// ---------------------------------------------------------------------------
__global__ __launch_bounds__(256) void final_kernel(
    const float* __restrict__ x, const float* __restrict__ z,
    const float* __restrict__ sc2, const float* __restrict__ sh2,
    float* __restrict__ out)
{
    const int idx = blockIdx.x * 256 + threadIdx.x;   // ((b*128+c)*128+h)*128+w
    const int c   = (idx >> 14) & 127;
    const int b   = idx >> 21;
    const int hw  = idx & (HW - 1);
    const int pix = (b << 14) + hw;
    const float zn = fmaf(z[(size_t)pix * CDIM + c], sc2[c], sh2[c]);
    const float v  = x[idx] + zn;
    out[idx] = 0.5f * v * (1.0f + erff(v * GELU_K));
}

// ---------------------------------------------------------------------------
// Launcher.
// ---------------------------------------------------------------------------
extern "C" void kernel_launch(void* const* d_in, const int* in_sizes, int n_in,
                              void* d_out, int out_size, void* d_ws, size_t ws_size,
                              hipStream_t stream)
{
    (void)in_sizes; (void)n_in; (void)out_size; (void)ws_size;

    const float* x    = (const float*)d_in[0];
    const float* offw = (const float*)d_in[1];
    const float* offb = (const float*)d_in[2];
    const float* defw = (const float*)d_in[3];
    const float* dww  = (const float*)d_in[4];
    const float* dwb  = (const float*)d_in[5];
    const float* dw2w = (const float*)d_in[6];
    const float* dw2b = (const float*)d_in[7];
    const float* bn1g = (const float*)d_in[8];
    const float* bn1b = (const float*)d_in[9];
    const float* w1   = (const float*)d_in[10];
    const float* b1   = (const float*)d_in[11];
    const float* w2   = (const float*)d_in[12];
    const float* b2   = (const float*)d_in[13];
    const float* bn2g = (const float*)d_in[14];
    const float* bn2b = (const float*)d_in[15];
    float* out = (float*)d_out;

    // Workspace layout (~129 MB):
    float*    yraw = (float*)d_ws;                         // 65536*128 f32
    float*    sc1  = yraw + (size_t)NPIX * CDIM;           // 128
    float*    sh1  = sc1 + 128;                            // 128
    _Float16* tbuf = (_Float16*)(sh1 + 128);               // 65536*512 f16
    float*    zbuf = (float*)(tbuf + (size_t)NPIX * HID);  // 65536*128 f32
    float*    sc2  = zbuf + (size_t)NPIX * CDIM;           // 128
    float*    sh2  = sc2 + 128;                            // 128

    spatial_kernel<<<NPIX / 256, 256, 0, stream>>>(
        x, offw, offb, defw, dww, dwb, dw2w, dw2b, yraw);
    bnstats_kernel<<<CDIM, 256, 0, stream>>>(yraw, CDIM, NPIX, bn1g, bn1b, sc1, sh1);
    gemm1_kernel<<<dim3(NPIX / 32, HID / 256), 256, 0, stream>>>(
        yraw, sc1, sh1, w1, b1, tbuf);
    gemm2_kernel<<<dim3(NPIX / 64, 1), 256, 0, stream>>>(tbuf, w2, b2, zbuf);
    bnstats_kernel<<<CDIM, 256, 0, stream>>>(zbuf, CDIM, NPIX, bn2g, bn2b, sc2, sh2);
    final_kernel<<<(NPIX * CDIM) / 256, 256, 0, stream>>>(x, zbuf, sc2, sh2, out);
}
